// BinaryMLP_44057774522909
// MI455X (gfx1250) — compile-verified, matrix operand order checked by reading, not run on hardware
//
#include <hip/hip_runtime.h>
#include <hip/hip_bf16.h>

// ---------------------------------------------------------------------------
// BinaryConnect MLP on gfx1250 via V_WMMA_F32_16X16X32_F16 + TDM staging.
//   h1 = relu(x @ sign(w1)^T + b1)      (65536x784)x(784x512)
//   h2 = relu(h1 @ sign(w2)^T + b2)     (65536x512)x(512x512)
//   h3 = relu(h2 @ sign(w3)^T + b3)     (65536x512)x(512x512)
//   y  = h3 @ w4^T + b4                 (65536x512)x(512x10)
// Binarized weights are exact in f16 (+-1); accumulate in f32 (WMMA C/D).
// Weight panel for each block is DMA'd to LDS once via TENSOR_LOAD_TO_LDS;
// the K-loop then feeds WMMA from ds_load_b128 (B) + global_load_b128 (A).
// ---------------------------------------------------------------------------

typedef _Float16 v16h __attribute__((ext_vector_type(16)));
typedef _Float16 v8h  __attribute__((ext_vector_type(8)));
typedef float    v8f  __attribute__((ext_vector_type(8)));
typedef float    v4f  __attribute__((ext_vector_type(4)));
typedef unsigned int u32x4 __attribute__((ext_vector_type(4)));
typedef int          i32x4 __attribute__((ext_vector_type(4)));
typedef int          i32x8 __attribute__((ext_vector_type(8)));

#define BATCH   65536
#define HID     512
#define K1      784
#define K1PAD   800   // w1b padded with zero columns so B loads need no guard

// ws layout (bytes)
#define OFF_W1B 0u
#define OFF_W2B (OFF_W1B + 512u * K1PAD * 2u)          //  819200
#define OFF_W3B (OFF_W2B + 512u * 512u * 2u)
#define OFF_W4B (OFF_W3B + 512u * 512u * 2u)
#define OFF_HA  (2u * 1024u * 1024u)                   // 2 MB aligned
#define OFF_HB  (OFF_HA + (size_t)BATCH * HID * 2u)    // +64 MB

// ---- Tensor Data Mover: stage `rows` x `elems` f16 panel (row stride
// `stride` elems) from global into LDS at the dynamic-shared base. ----------
__device__ __forceinline__ void stage_panel_to_lds(
    _Float16* smem, const _Float16* __restrict__ g,
    unsigned rows, unsigned elems, unsigned stride) {
#if defined(__gfx1250__) && __has_builtin(__builtin_amdgcn_tensor_load_to_lds)
  if ((threadIdx.x >> 5) == 0) {   // one TDM issue per workgroup (wave 0)
    unsigned long long ga = (unsigned long long)g;
    unsigned lds_off = (unsigned)__builtin_amdgcn_groupstaticsize();
    u32x4 g0;
    g0[0] = 1u;                                       // count=1 (valid user D#)
    g0[1] = lds_off;                                  // lds_addr (bytes)
    g0[2] = (unsigned)(ga & 0xffffffffu);             // global_addr[31:0]
    g0[3] = (unsigned)((ga >> 32) & 0x01ffffffu)      // global_addr[56:32]
          | (2u << 30);                               // type=2 ("image")
    i32x8 g1;
    g1[0] = (int)(1u << 16);                          // data_size=1 -> 2 bytes
    g1[1] = (int)((elems & 0xffffu) << 16);           // tensor_dim0[15:0]
    g1[2] = (int)(((elems >> 16) & 0xffffu)           // tensor_dim0[31:16]
          | ((rows & 0xffffu) << 16));                // tensor_dim1[15:0]
    g1[3] = (int)(((rows >> 16) & 0xffffu)            // tensor_dim1[31:16]
          | ((elems & 0xffffu) << 16));               // tile_dim0
    g1[4] = (int)(rows & 0xffffu);                    // tile_dim1 (tile_dim2=0)
    g1[5] = (int)stride;                              // tensor_dim0_stride[31:0]
    g1[6] = 0;                                        // stride hi / dim1_stride lo
    g1[7] = 0;
    i32x4 z4 = {0, 0, 0, 0};
    i32x8 z8 = {0, 0, 0, 0, 0, 0, 0, 0};
    __builtin_amdgcn_tensor_load_to_lds(g0, g1, z4, z4, z8, 0);
    __builtin_amdgcn_s_wait_tensorcnt((short)0);
  }
#else
  const unsigned chunks = (rows * elems) >> 3;        // v8h chunks
  for (unsigned i = threadIdx.x; i < chunks; i += blockDim.x) {
    unsigned r = i / (elems >> 3);
    unsigned c = (i % (elems >> 3)) << 3;
    *(v8h*)(smem + (size_t)r * elems + c) = *(const v8h*)(g + (size_t)r * stride + c);
  }
#endif
  __syncthreads();
}

// ---- fragment loaders -----------------------------------------------------
// 16-bit A/B fragment layout (16x32, K-major per lane):
//   lane l in [0,16):  row/col = l,     holds K = [0,8) and [16,24)
//   lane l in [16,32): row/col = l-16,  holds K = [8,16) and [24,32)
__device__ __forceinline__ v16h load_frag_f16(const _Float16* __restrict__ base,
                                              int ld, int lane) {
  const int h = lane >> 4;
  const int r = lane & 15;
  const _Float16* p = base + (size_t)r * ld + h * 8;
  v8h lo = *(const v8h*)(p);        // K chunk [h*8, h*8+8)
  v8h hi = *(const v8h*)(p + 16);   // K chunk [16+h*8, 16+h*8+8)
  return __builtin_shufflevector(lo, hi, 0,1,2,3,4,5,6,7,8,9,10,11,12,13,14,15);
}

// fp32 source with per-chunk K guard (K is a multiple of 8, so chunks are
// either fully in range or fully out).
__device__ __forceinline__ v16h load_frag_f32cvt(const float* __restrict__ rowbase,
                                                 int ld, int lane, int k0, int K) {
  const int h = lane >> 4;
  const int r = lane & 15;
  const float* p = rowbase + (size_t)r * ld + k0 + h * 8;
  v8h lo = (v8h)(_Float16)0;
  v8h hi = (v8h)(_Float16)0;
  if (k0 + h * 8 < K) {
    v4f a = *(const v4f*)(p);
    v4f b = *(const v4f*)(p + 4);
    v8f f = __builtin_shufflevector(a, b, 0,1,2,3,4,5,6,7);
    lo = __builtin_convertvector(f, v8h);
  }
  if (k0 + 16 + h * 8 < K) {
    v4f a = *(const v4f*)(p + 16);
    v4f b = *(const v4f*)(p + 20);
    v8f f = __builtin_shufflevector(a, b, 0,1,2,3,4,5,6,7);
    hi = __builtin_convertvector(f, v8h);
  }
  return __builtin_shufflevector(lo, hi, 0,1,2,3,4,5,6,7,8,9,10,11,12,13,14,15);
}

__device__ __forceinline__ v8f wmma_f16(v16h a, v16h b, v8f c) {
  return __builtin_amdgcn_wmma_f32_16x16x32_f16(false, a, false, b,
                                                (short)0, c, false, false);
}

// ---- weight prep: binarize/convert into ws --------------------------------
__global__ __launch_bounds__(256) void prep_weights(
    const float* __restrict__ w1, const float* __restrict__ w2,
    const float* __restrict__ w3, const float* __restrict__ w4,
    _Float16* __restrict__ w1b, _Float16* __restrict__ w2b,
    _Float16* __restrict__ w3b, _Float16* __restrict__ w4b) {
  const unsigned N1 = 512u * K1PAD;          // 409600
  const unsigned N2 = 512u * 512u;           // 262144
  const unsigned N4 = 16u * 512u;            // 8192
  unsigned i = blockIdx.x * 256u + threadIdx.x;
  if (i < N1) {
    unsigned n = i / K1PAD, k = i % K1PAD;
    _Float16 v = (_Float16)0;
    if (k < K1) v = (w1[(size_t)n * K1 + k] >= 0.0f) ? (_Float16)1.0f : (_Float16)-1.0f;
    w1b[i] = v;
  } else if (i < N1 + N2) {
    unsigned j = i - N1;
    w2b[j] = (w2[j] >= 0.0f) ? (_Float16)1.0f : (_Float16)-1.0f;
  } else if (i < N1 + 2u * N2) {
    unsigned j = i - N1 - N2;
    w3b[j] = (w3[j] >= 0.0f) ? (_Float16)1.0f : (_Float16)-1.0f;
  } else if (i < N1 + 2u * N2 + N4) {
    unsigned j = i - N1 - 2u * N2;
    unsigned n = j / 512u;
    w4b[j] = (n < 10u) ? (_Float16)w4[j] : (_Float16)0.0f;
  }
}

// ---- binarized linear + bias + relu ---------------------------------------
// Block: 256 threads = 8 waves arranged 4(M) x 2(N); block tile 128x128.
// Wave tile 32x64 -> 2x4 grid of 16x16 accumulators (8 x v8f = 64 VGPRs).
// The 128 x ldb weight panel for this block is TDM-staged to LDS up front.
template <bool A_IS_F32>
__global__ __launch_bounds__(256) void gemm_bin_relu(
    const void* __restrict__ Aptr,          // M x lda (f32 or f16), row-major
    const _Float16* __restrict__ B,         // N x ldb row-major (+-1 f16, padded)
    const float* __restrict__ bias,         // N
    _Float16* __restrict__ C,               // M x N row-major f16
    int N, int K, int lda, int ldb) {
  extern __shared__ _Float16 smemB[];       // 128 * ldb f16
  const int lane  = threadIdx.x & 31;
  const int wave  = threadIdx.x >> 5;
  const int m0    = blockIdx.x * 128 + (wave >> 1) * 32;
  const int nloc  = (wave & 1) * 64;        // wave's N offset inside block
  const int n0    = blockIdx.y * 128 + nloc;

  stage_panel_to_lds(smemB, B + (size_t)blockIdx.y * 128 * ldb, 128, ldb, ldb);

  v8f acc[2][4];
#pragma unroll
  for (int i = 0; i < 2; ++i)
#pragma unroll
    for (int j = 0; j < 4; ++j) acc[i][j] = (v8f)0.0f;

  const int K32 = (K + 31) >> 5;
  for (int kk = 0; kk < K32; ++kk) {
    const int k0 = kk * 32;
    v16h a[2], b[4];
    if (A_IS_F32) {
      const float* A = (const float*)Aptr;
      a[0] = load_frag_f32cvt(A + (size_t)m0 * lda,        lda, lane, k0, K);
      a[1] = load_frag_f32cvt(A + (size_t)(m0 + 16) * lda, lda, lane, k0, K);
    } else {
      const _Float16* A = (const _Float16*)Aptr;
      a[0] = load_frag_f16(A + (size_t)m0 * lda + k0,        lda, lane);
      a[1] = load_frag_f16(A + (size_t)(m0 + 16) * lda + k0, lda, lane);
    }
#pragma unroll
    for (int j = 0; j < 4; ++j)   // B fragments from LDS (ds_load_b128)
      b[j] = load_frag_f16(smemB + (size_t)(nloc + j * 16) * ldb + k0, ldb, lane);
#pragma unroll
    for (int i = 0; i < 2; ++i)
#pragma unroll
      for (int j = 0; j < 4; ++j)
        acc[i][j] = wmma_f16(a[i], b[j], acc[i][j]);
  }

  // Epilogue: C/D layout -> lane l holds col n0+j*16+(l&15), rows m0+(l>>4)*8+r.
  const int h  = lane >> 4;
  const int nl = lane & 15;
#pragma unroll
  for (int j = 0; j < 4; ++j) {
    const int n = n0 + j * 16 + nl;
    const float bs = bias[n];
#pragma unroll
    for (int i = 0; i < 2; ++i) {
      const int mbase = m0 + i * 16 + h * 8;
#pragma unroll
      for (int r = 0; r < 8; ++r) {
        float v = acc[i][j][r] + bs;
        v = v > 0.0f ? v : 0.0f;
        C[(size_t)(mbase + r) * N + n] = (_Float16)v;
      }
    }
  }
}

// ---- final fp32 layer: (M x 512) x (512 x 16pad), store cols 0..9 ---------
// Block: 256 threads = 8 waves, each wave owns 32 rows; block tile 256 x 16.
__global__ __launch_bounds__(256) void gemm_out(
    const _Float16* __restrict__ A,   // M x 512 f16
    const _Float16* __restrict__ B,   // 16 x 512 f16 (rows >=10 are zero)
    const float* __restrict__ bias,   // 10
    float* __restrict__ out) {        // M x 10 f32
  extern __shared__ _Float16 smemB[];       // 16 x 512 f16
  const int lane = threadIdx.x & 31;
  const int wave = threadIdx.x >> 5;
  const int m0   = blockIdx.x * 256 + wave * 32;

  stage_panel_to_lds(smemB, B, 16, HID, HID);

  v8f acc[2];
  acc[0] = (v8f)0.0f;
  acc[1] = (v8f)0.0f;

  for (int kk = 0; kk < HID / 32; ++kk) {
    const int k0 = kk * 32;
    v16h a0 = load_frag_f16(A + (size_t)m0 * HID + k0,        HID, lane);
    v16h a1 = load_frag_f16(A + (size_t)(m0 + 16) * HID + k0, HID, lane);
    v16h b  = load_frag_f16(smemB + k0, HID, lane);
    acc[0] = wmma_f16(a0, b, acc[0]);
    acc[1] = wmma_f16(a1, b, acc[1]);
  }

  const int h = lane >> 4;
  const int n = lane & 15;
  if (n < 10) {
    const float bs = bias[n];
#pragma unroll
    for (int i = 0; i < 2; ++i) {
      const int mbase = m0 + i * 16 + h * 8;
#pragma unroll
      for (int r = 0; r < 8; ++r)
        out[(size_t)(mbase + r) * 10 + n] = acc[i][r] + bs;
    }
  }
}

// ---------------------------------------------------------------------------
extern "C" void kernel_launch(void* const* d_in, const int* in_sizes, int n_in,
                              void* d_out, int out_size, void* d_ws, size_t ws_size,
                              hipStream_t stream) {
  const float* x  = (const float*)d_in[0];
  const float* w1 = (const float*)d_in[1];
  const float* b1 = (const float*)d_in[2];
  const float* w2 = (const float*)d_in[3];
  const float* b2 = (const float*)d_in[4];
  const float* w3 = (const float*)d_in[5];
  const float* b3 = (const float*)d_in[6];
  const float* w4 = (const float*)d_in[7];
  const float* b4 = (const float*)d_in[8];

  char* ws = (char*)d_ws;
  _Float16* w1b = (_Float16*)(ws + OFF_W1B);
  _Float16* w2b = (_Float16*)(ws + OFF_W2B);
  _Float16* w3b = (_Float16*)(ws + OFF_W3B);
  _Float16* w4b = (_Float16*)(ws + OFF_W4B);
  _Float16* hA  = (_Float16*)(ws + OFF_HA);
  _Float16* hB  = (_Float16*)(ws + OFF_HB);

  // 1) binarize / convert weights
  {
    const unsigned total = 512u * K1PAD + 2u * 512u * 512u + 16u * 512u; // 942080
    prep_weights<<<(total + 255u) / 256u, 256, 0, stream>>>(w1, w2, w3, w4,
                                                            w1b, w2b, w3b, w4b);
  }

  // 2) fc1: x(f32) -> hA, K=784 (B padded to 800); LDS panel = 128*800*2 B
  gemm_bin_relu<true><<<dim3(BATCH / 128, HID / 128), 256,
                        128 * K1PAD * 2, stream>>>(
      x, w1b, b1, hA, HID, K1, K1, K1PAD);

  // 3) fc2: hA -> hB; LDS panel = 128*512*2 B
  gemm_bin_relu<false><<<dim3(BATCH / 128, HID / 128), 256,
                         128 * HID * 2, stream>>>(
      hA, w2b, b2, hB, HID, HID, HID, HID);

  // 4) fc3: hB -> hA
  gemm_bin_relu<false><<<dim3(BATCH / 128, HID / 128), 256,
                         128 * HID * 2, stream>>>(
      hB, w3b, b3, hA, HID, HID, HID, HID);

  // 5) fc4: hA -> out (fp32, N=10); LDS panel = 16*512*2 B
  gemm_out<<<BATCH / 256, 256, 16 * HID * 2, stream>>>(hA, w4b, b4,
                                                       (float*)d_out);
}
